// CriticNetwork_59425167507826
// MI455X (gfx1250) — matrix operand order repaired and checked by reading, not compiled, not run on hardware
//
#include <hip/hip_runtime.h>

typedef _Float16 half_t;
typedef __attribute__((ext_vector_type(16))) _Float16 v16h;
typedef __attribute__((ext_vector_type(8)))  _Float16 v8h;
typedef __attribute__((ext_vector_type(8)))  float    v8f;

#define BB   128
#define NN   32
#define BN   (BB*NN)        // 4096
#define BNN  (BB*NN*NN)     // 131072

// ---------------------------------------------------------------------------
// WMMA fragment loaders (16x16x32 f16, wave32; this code's contiguous k order, same for A and B)
// A (16x32, row-major): lane&15 = M; lanes 16..31 hold K+16; halves 0..15 =
//   K[c0..c0+15] contiguous
// B (32x16): lane&15 = N col; lanes 16..31 hold K+16; 16 contiguous K halves
// C/D (16x16 f32): lane&15 = N; vgpr j -> M = (lane>>4)*8 + j
// ---------------------------------------------------------------------------
__device__ inline v16h load_a_frag(const half_t* __restrict__ X, int ldk, int kc, int lane) {
    int r  = lane & 15;
    int c0 = kc + ((lane >> 4) << 4);
    v8h lo = *(const v8h*)(X + r * ldk + c0);
    v8h hi = *(const v8h*)(X + r * ldk + c0 + 8);
    v16h a;
#pragma unroll
    for (int i = 0; i < 8; ++i) { a[i] = lo[i]; a[8 + i] = hi[i]; }
    return a;
}

__device__ inline v16h load_b_frag(const half_t* __restrict__ WT, int ldk, int col_base, int kc, int lane) {
    int n  = col_base + (lane & 15);
    int kb = kc + ((lane >> 4) << 4);
    return *(const v16h*)(WT + n * ldk + kb);
}

__device__ inline float fast_rcp(float x) { return __builtin_amdgcn_rcpf(x); }

// ---------------------------------------------------------------------------
// Fused 2-layer MLP: Y16[rows][64] = leaky(X16[rows][KPAD] @ W1 + b1) @ W2 + b2
// 8 waves/block, one 16-row tile per wave; hidden tile staged in LDS.
// ---------------------------------------------------------------------------
template <int KPAD>
__global__ __launch_bounds__(256) void mlp2(const half_t* __restrict__ X,
                                            const half_t* __restrict__ W1T,
                                            const float*  __restrict__ b1,
                                            const half_t* __restrict__ W2T,
                                            const float*  __restrict__ b2,
                                            half_t* __restrict__ Y) {
    __shared__ alignas(32) half_t sW1[64 * KPAD];
    __shared__ alignas(32) half_t sW2[64 * 64];
    __shared__ alignas(32) half_t sH[8 * 16 * 64];
    {
        const uint32_t* s1 = (const uint32_t*)W1T;
        uint32_t*       d1 = (uint32_t*)sW1;
        for (int i = threadIdx.x; i < 64 * KPAD / 2; i += 256) d1[i] = s1[i];
        const uint32_t* s2 = (const uint32_t*)W2T;
        uint32_t*       d2 = (uint32_t*)sW2;
        for (int i = threadIdx.x; i < 64 * 64 / 2; i += 256) d2[i] = s2[i];
    }
    const int wave    = threadIdx.x >> 5;
    const int lane    = threadIdx.x & 31;
    const int rowBase = (blockIdx.x * 8 + wave) * 16;
    const half_t* Xr  = X + (size_t)rowBase * KPAD;
    __builtin_prefetch(Xr + (lane & 15) * KPAD + ((lane >> 4) << 6), 0, 3); // global_prefetch_b8 (near)
    __syncthreads();

    // ---- layer 1 ----
    v8f acc[4] = {};
#pragma unroll
    for (int kc = 0; kc < KPAD; kc += 32) {
        v16h a = load_a_frag(Xr, KPAD, kc, lane);
#pragma unroll
        for (int nt = 0; nt < 4; ++nt) {
            v16h b = load_b_frag(sW1, KPAD, nt * 16, kc, lane);
            acc[nt] = __builtin_amdgcn_wmma_f32_16x16x32_f16(
                false, a, false, b, (short)0, acc[nt], false, false);
        }
    }
    half_t* hMine = sH + wave * (16 * 64);
#pragma unroll
    for (int nt = 0; nt < 4; ++nt) {
        int   n  = nt * 16 + (lane & 15);
        float bv = b1[n];
#pragma unroll
        for (int j = 0; j < 8; ++j) {
            int   m = ((lane >> 4) << 3) + j;
            float v = acc[nt][j] + bv;
            v = v > 0.f ? v : 0.01f * v;
            hMine[m * 64 + n] = (half_t)v;
        }
    }
    __syncthreads();

    // ---- layer 2 (hidden tile from LDS) ----
    v8f acc2[4] = {};
#pragma unroll
    for (int kc = 0; kc < 64; kc += 32) {
        v16h a = load_a_frag(hMine, 64, kc, lane);
#pragma unroll
        for (int nt = 0; nt < 4; ++nt) {
            v16h b = load_b_frag(sW2, 64, nt * 16, kc, lane);
            acc2[nt] = __builtin_amdgcn_wmma_f32_16x16x32_f16(
                false, a, false, b, (short)0, acc2[nt], false, false);
        }
    }
#pragma unroll
    for (int nt = 0; nt < 4; ++nt) {
        int   n  = nt * 16 + (lane & 15);
        float bv = b2[n];
#pragma unroll
        for (int j = 0; j < 8; ++j) {
            int m = ((lane >> 4) << 3) + j;
            Y[(size_t)(rowBase + m) * 64 + n] = (half_t)(acc2[nt][j] + bv);
        }
    }
}

// ---------------------------------------------------------------------------
// Fused fv MLP + 64->1 head:
//   value[row] = leaky(node_in[row][128] @ fv1 + b1) . fv2_w + fv2_b
// ---------------------------------------------------------------------------
__global__ __launch_bounds__(256) void fv_head(const half_t* __restrict__ X,
                                               const half_t* __restrict__ W1T,
                                               const float*  __restrict__ b1,
                                               const float*  __restrict__ w2,
                                               const float*  __restrict__ b2,
                                               float* __restrict__ out) {
    constexpr int KPAD = 128;
    __shared__ alignas(32) half_t sW1[64 * KPAD];
    {
        const uint32_t* s1 = (const uint32_t*)W1T;
        uint32_t*       d1 = (uint32_t*)sW1;
        for (int i = threadIdx.x; i < 64 * KPAD / 2; i += 256) d1[i] = s1[i];
    }
    const int wave    = threadIdx.x >> 5;
    const int lane    = threadIdx.x & 31;
    const int rowBase = (blockIdx.x * 8 + wave) * 16;
    const half_t* Xr  = X + (size_t)rowBase * KPAD;
    __builtin_prefetch(Xr + (lane & 15) * KPAD + ((lane >> 4) << 6), 0, 3);
    __syncthreads();

    v8f acc[4] = {};
#pragma unroll
    for (int kc = 0; kc < KPAD; kc += 32) {
        v16h a = load_a_frag(Xr, KPAD, kc, lane);
#pragma unroll
        for (int nt = 0; nt < 4; ++nt) {
            v16h b = load_b_frag(sW1, KPAD, nt * 16, kc, lane);
            acc[nt] = __builtin_amdgcn_wmma_f32_16x16x32_f16(
                false, a, false, b, (short)0, acc[nt], false, false);
        }
    }
    float psum[8] = {};
#pragma unroll
    for (int nt = 0; nt < 4; ++nt) {
        int   n  = nt * 16 + (lane & 15);
        float bv = b1[n];
        float wv = w2[n];
#pragma unroll
        for (int j = 0; j < 8; ++j) {
            float v = acc[nt][j] + bv;
            v = v > 0.f ? v : 0.01f * v;
            psum[j] += v * wv;
        }
    }
#pragma unroll
    for (int j = 0; j < 8; ++j) {
#pragma unroll
        for (int o = 8; o > 0; o >>= 1) psum[j] += __shfl_xor(psum[j], o, 32);
    }
    if ((lane & 15) == 0) {                      // lane 0 -> rows 0..7, lane 16 -> rows 8..15
        float bb = b2[0];
        int   mb = (lane >> 4) << 3;
#pragma unroll
        for (int j = 0; j < 8; ++j) out[rowBase + mb + j] = psum[j] + bb;
    }
}

// ---------------------------------------------------------------------------
// scores_z[b,a,j] = key_z[b,a,:].query_z[b,j,:];  weight_z = sigmoid(s/8)
// ---------------------------------------------------------------------------
__global__ __launch_bounds__(128) void scores_z_kernel(const half_t* __restrict__ keyz,
                                                       const half_t* __restrict__ qz,
                                                       float* __restrict__ out_wz) {
    const int b = blockIdx.x, wave = threadIdx.x >> 5, lane = threadIdx.x & 31;
    const int mt = wave >> 1, nt = wave & 1;
    const half_t* K = keyz + (size_t)(b * 32 + mt * 16) * 64;
    v8f acc = {};
#pragma unroll
    for (int kc = 0; kc < 64; kc += 32) {
        v16h a  = load_a_frag(K, 64, kc, lane);
        v16h bf = *(const v16h*)(qz + (size_t)(b * 32 + nt * 16 + (lane & 15)) * 64
                                 + kc + ((lane >> 4) << 4));
        acc = __builtin_amdgcn_wmma_f32_16x16x32_f16(false, a, false, bf, (short)0, acc, false, false);
    }
#pragma unroll
    for (int j = 0; j < 8; ++j) {
        int   a_i = mt * 16 + ((lane >> 4) << 3) + j;
        int   n   = nt * 16 + (lane & 15);
        float s   = acc[j] * 0.125f;
        out_wz[b * 1024 + a_i * 32 + n] = fast_rcp(1.f + __expf(-s));
    }
}

// ---------------------------------------------------------------------------
// glue kernels (init16/obsz/nodein row stores vectorized to b128)
// ---------------------------------------------------------------------------
__device__ inline v8h cvt8(float4 f0, float4 f1) {
    v8h h = { (half_t)f0.x, (half_t)f0.y, (half_t)f0.z, (half_t)f0.w,
              (half_t)f1.x, (half_t)f1.y, (half_t)f1.z, (half_t)f1.w };
    return h;
}

__global__ void wconv_kernel(const float* __restrict__ W, half_t* __restrict__ WT,
                             int din, int dout, int kpad) {
    int idx = blockIdx.x * blockDim.x + threadIdx.x;
    if (idx >= dout * kpad) return;
    int o = idx / kpad, k = idx % kpad;
    WT[o * kpad + k] = (half_t)((k < din) ? W[k * dout + o] : 0.f);
}

__global__ void init16_kernel(const float* __restrict__ states, const float* __restrict__ policies,
                              half_t* __restrict__ s16, half_t* __restrict__ sp16) {
    int row = blockIdx.x * blockDim.x + threadIdx.x;   // one row per thread
    if (row >= BN) return;
    const float4* st = (const float4*)(states   + (size_t)row * 64);
    const float4* po = (const float4*)(policies + (size_t)row * 16);
    half_t* s  = s16  + (size_t)row * 64;
    half_t* sp = sp16 + (size_t)row * 96;
#pragma unroll
    for (int c8 = 0; c8 < 8; ++c8) {
        v8h h = cvt8(st[2 * c8], st[2 * c8 + 1]);
        *(v8h*)(s + 8 * c8)  = h;
        *(v8h*)(sp + 8 * c8) = h;
    }
#pragma unroll
    for (int k8 = 0; k8 < 2; ++k8)
        *(v8h*)(sp + 64 + 8 * k8) = cvt8(po[2 * k8], po[2 * k8 + 1]);
    v8h z = {};
    *(v8h*)(sp + 80) = z;
    *(v8h*)(sp + 88) = z;
}

__global__ void obsz_kernel(const float* __restrict__ states, const float* __restrict__ actions,
                            const float* __restrict__ policies, const float* __restrict__ wz,
                            half_t* __restrict__ obsz, half_t* __restrict__ src16) {
    int row = blockIdx.x * blockDim.x + threadIdx.x;   // (b,a,j)
    if (row >= BNN) return;
    int b = row >> 10, a = (row >> 5) & 31, j = row & 31;
    float w = wz[row];
    const float4* st = (const float4*)(states   + (size_t)(b * 32 + j) * 64);
    const float4* ac = (const float4*)(actions  + (size_t)(b * 32 + j) * 16);
    const float4* po = (const float4*)(policies + (size_t)(b * 32 + j) * 16);
    half_t* dst = obsz + (size_t)row * 96;
    const bool diag = (a == j);
    half_t* sd = src16 + (size_t)(b * 32 + a) * 96;
#pragma unroll
    for (int c8 = 0; c8 < 8; ++c8) {
        v8h h = cvt8(st[2 * c8], st[2 * c8 + 1]);
        *(v8h*)(dst + 8 * c8) = h;
        if (diag) *(v8h*)(sd + 8 * c8) = h;
    }
#pragma unroll
    for (int k8 = 0; k8 < 2; ++k8) {
        float4 a0 = ac[2 * k8], a1 = ac[2 * k8 + 1];
        float4 p0 = po[2 * k8], p1 = po[2 * k8 + 1];
        v8h h = { (half_t)(w * a0.x + (1.f - w) * p0.x), (half_t)(w * a0.y + (1.f - w) * p0.y),
                  (half_t)(w * a0.z + (1.f - w) * p0.z), (half_t)(w * a0.w + (1.f - w) * p0.w),
                  (half_t)(w * a1.x + (1.f - w) * p1.x), (half_t)(w * a1.y + (1.f - w) * p1.y),
                  (half_t)(w * a1.z + (1.f - w) * p1.z), (half_t)(w * a1.w + (1.f - w) * p1.w) };
        *(v8h*)(dst + 64 + 8 * k8) = h;
        if (diag) *(v8h*)(sd + 64 + 8 * k8) = h;
    }
    v8h z = {};
    *(v8h*)(dst + 80) = z;
    *(v8h*)(dst + 88) = z;
    if (diag) { *(v8h*)(sd + 80) = z; *(v8h*)(sd + 88) = z; }
}

// softmax over j of  key_obsz[b,a,:].query_obsz[b,a,j,:]/8 ; one wave per (b,a)
__global__ __launch_bounds__(128) void softmax_kernel(const half_t* __restrict__ ko,
                                                      const half_t* __restrict__ qo,
                                                      float* __restrict__ out_wo,
                                                      float* __restrict__ ws_wo) {
    int wid  = (blockIdx.x * blockDim.x + threadIdx.x) >> 5;  // (b,a)
    int lane = threadIdx.x & 31;                              // j
    if (wid >= BN) return;
    const v8h* kr = (const v8h*)(ko + (size_t)wid * 64);
    const v8h* qr = (const v8h*)(qo + ((size_t)wid * 32 + lane) * 64);
    float s = 0.f;
#pragma unroll
    for (int d8 = 0; d8 < 8; ++d8) {
        v8h k = kr[d8], q = qr[d8];
#pragma unroll
        for (int i = 0; i < 8; ++i) s += (float)k[i] * (float)q[i];
    }
    s *= 0.125f;
    float m = s;
#pragma unroll
    for (int o = 16; o > 0; o >>= 1) m = fmaxf(m, __shfl_xor(m, o, 32));
    float e = __expf(s - m), sum = e;
#pragma unroll
    for (int o = 16; o > 0; o >>= 1) sum += __shfl_xor(sum, o, 32);
    float w = e * fast_rcp(sum);
    out_wo[(size_t)wid * 32 + lane] = w;
    ws_wo [(size_t)wid * 32 + lane] = w;
}

// sum_all[b,a,d] = sum_j w[b,a,j] * vals[b,a,j,d] ; one wave per (b,a)
__global__ __launch_bounds__(128) void sumall_kernel(const float* __restrict__ wo,
                                                     const half_t* __restrict__ vals,
                                                     float* __restrict__ sumall) {
    int wid  = (blockIdx.x * blockDim.x + threadIdx.x) >> 5;
    int lane = threadIdx.x & 31;
    if (wid >= BN) return;
    float a0 = 0.f, a1 = 0.f;
    for (int j = 0; j < 32; ++j) {
        float w = wo[(size_t)wid * 32 + j];
        const half_t* v = vals + ((size_t)wid * 32 + j) * 64;
        a0 += w * (float)v[lane];
        a1 += w * (float)v[lane + 32];
    }
    sumall[(size_t)wid * 64 + lane]      = a0;
    sumall[(size_t)wid * 64 + 32 + lane] = a1;
}

__global__ void nodein_kernel(const half_t* __restrict__ s16, const float* __restrict__ wo,
                              const float* __restrict__ sumall, const half_t* __restrict__ vals,
                              const half_t* __restrict__ valspi, half_t* __restrict__ nodein) {
    int row = blockIdx.x * blockDim.x + threadIdx.x;   // (b,a,j)
    if (row >= BNN) return;
    int b = row >> 10, a = (row >> 5) & 31, j = row & 31;
    const v8h*   st = (const v8h*)(s16    + (size_t)(b * 32 + j) * 64);
    const float* sa = sumall + (size_t)(b * 32 + a) * 64;
    const v8h*   vv = (const v8h*)(vals   + (size_t)row * 64);
    const v8h*   vp = (const v8h*)(valspi + (size_t)(b * 32 + j) * 64);
    float w = wo[row];
    half_t* dst = nodein + (size_t)row * 128;
#pragma unroll
    for (int c8 = 0; c8 < 8; ++c8) *(v8h*)(dst + 8 * c8) = st[c8];
#pragma unroll
    for (int d8 = 0; d8 < 8; ++d8) {
        v8h a8 = vv[d8], p8 = vp[d8], h;
#pragma unroll
        for (int i = 0; i < 8; ++i)
            h[i] = (half_t)((sa[8 * d8 + i] - w * (float)a8[i] + w * (float)p8[i]) * (1.f / 32.f));
        *(v8h*)(dst + 64 + 8 * d8) = h;
    }
}

// ---------------------------------------------------------------------------
extern "C" void kernel_launch(void* const* d_in, const int* in_sizes, int n_in,
                              void* d_out, int out_size, void* d_ws, size_t ws_size,
                              hipStream_t stream) {
    const float* states   = (const float*)d_in[0];
    const float* policies = (const float*)d_in[1];
    const float* actions  = (const float*)d_in[2];
    const float *W[12], *Bp[12];   // kw1,kw2,qw1,qw2,ko1,ko2,qo1,qo2,av1,av2,fv1,fv2
    for (int s = 0; s < 6; ++s) {
        W [2*s]   = (const float*)d_in[3 + 4*s + 0];
        Bp[2*s]   = (const float*)d_in[3 + 4*s + 1];
        W [2*s+1] = (const float*)d_in[3 + 4*s + 2];
        Bp[2*s+1] = (const float*)d_in[3 + 4*s + 3];
    }
    float* out_value = (float*)d_out;             // (B,N,N,1)
    float* out_wz    = (float*)d_out + BNN;       // (B,N,N)
    float* out_wo    = (float*)d_out + 2 * BNN;   // (B,N,N,1)

    // ---- workspace layout ----
    char*  ws  = (char*)d_ws;
    size_t off = 0;
    auto alloc = [&](size_t bytes) { size_t o = off; off = (off + bytes + 255) & ~(size_t)255; return ws + o; };

    const int din[11]  = {64,64, 64,64, 80,64, 80,64, 80,64, 128};
    const int kpad[11] = {64,64, 64,64, 96,64, 96,64, 96,64, 128};
    half_t* wt[11];
    for (int i = 0; i < 11; ++i) wt[i] = (half_t*)alloc((size_t)64 * kpad[i] * 2);

    half_t* s16      = (half_t*)alloc((size_t)BN  * 64  * 2);
    half_t* sp16     = (half_t*)alloc((size_t)BN  * 96  * 2);
    half_t* keyz16   = (half_t*)alloc((size_t)BN  * 64  * 2);
    half_t* qz16     = (half_t*)alloc((size_t)BN  * 64  * 2);
    half_t* big      = (half_t*)alloc((size_t)BNN * 128 * 2); // obs_z (96) then node_in (128)
    half_t* obsz16   = big;
    half_t* nodein16 = big;
    half_t* src16    = (half_t*)alloc((size_t)BN  * 96  * 2);
    half_t* kobs16   = (half_t*)alloc((size_t)BN  * 64  * 2);
    half_t* qobs16   = (half_t*)alloc((size_t)BNN * 64  * 2);
    half_t* vals16   = (half_t*)alloc((size_t)BNN * 64  * 2);
    half_t* valspi16 = (half_t*)alloc((size_t)BN  * 64  * 2);
    float*  wobsz    = (float*)alloc((size_t)BN * 32 * 4);
    float*  sumall   = (float*)alloc((size_t)BN * 64 * 4);

    // ---- 1: weights -> transposed, K-padded f16; inputs -> f16 ----
    for (int i = 0; i < 11; ++i) {
        int n = 64 * kpad[i];
        wconv_kernel<<<(n + 255) / 256, 256, 0, stream>>>(W[i], wt[i], din[i], 64, kpad[i]);
    }
    init16_kernel<<<BN / 256, 256, 0, stream>>>(states, policies, s16, sp16);

    // ---- 2: key_z / query_z fused MLPs (WMMA x2 layers, LDS hidden) ----
    mlp2<64><<<BN / 128, 256, 0, stream>>>(s16, wt[0], Bp[0], wt[1], Bp[1], keyz16);
    mlp2<64><<<BN / 128, 256, 0, stream>>>(s16, wt[2], Bp[2], wt[3], Bp[3], qz16);

    // ---- 3: attention-z scores + sigmoid -> weight_z (WMMA) ----
    scores_z_kernel<<<BB, 128, 0, stream>>>(keyz16, qz16, out_wz);

    // ---- 4: build obs_z (B,N,N,96) and its diagonal ----
    obsz_kernel<<<BNN / 256, 256, 0, stream>>>(states, actions, policies, out_wz, obsz16, src16);

    // ---- 5: ko / qo / av fused MLPs (WMMA) ----
    mlp2<96><<<BN  / 128, 256, 0, stream>>>(src16,  wt[4], Bp[4], wt[5], Bp[5], kobs16);
    mlp2<96><<<BNN / 128, 256, 0, stream>>>(obsz16, wt[6], Bp[6], wt[7], Bp[7], qobs16);
    mlp2<96><<<BNN / 128, 256, 0, stream>>>(obsz16, wt[8], Bp[8], wt[9], Bp[9], vals16);
    mlp2<96><<<BN  / 128, 256, 0, stream>>>(sp16,   wt[8], Bp[8], wt[9], Bp[9], valspi16);

    // ---- 6: softmax attention weights + weighted sums ----
    softmax_kernel<<<BN / 4, 128, 0, stream>>>(kobs16, qobs16, out_wo, wobsz);
    sumall_kernel <<<BN / 4, 128, 0, stream>>>(wobsz, vals16, sumall);

    // ---- 7: node features -> fused fv MLP + value head (WMMA + shfl head) ----
    nodein_kernel<<<BNN / 256, 256, 0, stream>>>(s16, wobsz, sumall, vals16, valspi16, nodein16);
    fv_head<<<BNN / 128, 256, 0, stream>>>(nodein16, wt[10], Bp[10], W[11], Bp[11], out_value);
    (void)in_sizes; (void)n_in; (void)out_size; (void)ws_size;
}
